// RelationAttentionMLPHead_10539849744626
// MI455X (gfx1250) — matrix operand order repaired and checked by reading, not compiled, not run on hardware
//
#include <hip/hip_runtime.h>
#include <hip/hip_bf16.h>

typedef __attribute__((ext_vector_type(16))) __bf16 v16bf;
typedef __attribute__((ext_vector_type(2)))  __bf16 v2bf;
typedef __attribute__((ext_vector_type(8)))  float  v8f;

// f32 -> bf16 via native fptrunc (RTNE); lets the backend pick the HW cvt op.
__device__ __forceinline__ unsigned short f2bfu(float f) {
    __bf16 h = (__bf16)f;
    return __builtin_bit_cast(unsigned short, h);
}
__device__ __forceinline__ float bfu2f(unsigned short h) {
    unsigned u = ((unsigned)h) << 16;
    return __builtin_bit_cast(float, u);
}
// packed pair convert (vectorizable into v_cvt_pk form if available)
__device__ __forceinline__ unsigned cvt2(float lo, float hi) {
    v2bf p;
    p[0] = (__bf16)lo;
    p[1] = (__bf16)hi;
    return __builtin_bit_cast(unsigned, p);
}

union ABu {
    v16bf v;
    unsigned short us[16];
    unsigned u32[8];
    uint4 q[2];
};

#define WMMA_BF16(A, B, C) \
    __builtin_amdgcn_wmma_f32_16x16x32_bf16(false, (A), false, (B), (short)0, (C), false, false)

// A fragment from a global f32 row (16 values -> bf16)
__device__ __forceinline__ ABu mk_afrag_g(const float* rp) {
    float4 x0 = *(const float4*)rp;
    float4 x1 = *(const float4*)(rp + 4);
    float4 x2 = *(const float4*)(rp + 16);
    float4 x3 = *(const float4*)(rp + 20);
    ABu a;
    a.u32[0] = cvt2(x0.x, x0.y); a.u32[1] = cvt2(x0.z, x0.w);
    a.u32[2] = cvt2(x1.x, x1.y); a.u32[3] = cvt2(x1.z, x1.w);
    a.u32[4] = cvt2(x2.x, x2.y); a.u32[5] = cvt2(x2.z, x2.w);
    a.u32[6] = cvt2(x3.x, x3.y); a.u32[7] = cvt2(x3.z, x3.w);
    return a;
}
// A fragment from an LDS bf16 row
__device__ __forceinline__ ABu mk_afrag_l(const unsigned short* row, int k0) {
    ABu a;
    a.q[0] = *(const uint4*)(row + k0);
    a.q[1] = *(const uint4*)(row + k0 + 16);
    return a;
}

// ---------------------------------------------------------------------------
// Prep kernel 1: Q_table[10][128] = rel_emb @ W_q + b_q (only 10 distinct rows)
// ---------------------------------------------------------------------------
__global__ void build_qtab(const float* __restrict__ rel_emb,
                           const float* __restrict__ W_q,
                           const float* __restrict__ b_q,
                           float* __restrict__ qtab) {
    int n = threadIdx.x; // 0..127
    for (int r = 0; r < 10; ++r) {
        float s = b_q[n];
        for (int d = 0; d < 64; ++d)
            s += rel_emb[r * 64 + d] * W_q[d * 128 + n];
        qtab[r * 128 + n] = s;
    }
}

// ---------------------------------------------------------------------------
// Prep kernel 2: pack f32 row-major [K][128] weight into bf16 WMMA B-fragment
// layout: per (kTile,nTile): 32 lanes x 16 bf16 contiguous.
// Lane l holds column n = nTile*16 + (l&15), K rows kTile*32 + (l>>4)*8 + kk (kk<8), +16 for kk>=8.
// ---------------------------------------------------------------------------
__global__ void pack_b(const float* __restrict__ W, int kTiles,
                       unsigned short* __restrict__ outp) {
    const int N = 128;
    int t = blockIdx.x * blockDim.x + threadIdx.x;
    int total = kTiles * (N / 16) * 32;
    if (t >= total) return;
    int lane = t & 31;
    int rest = t >> 5;
    int nt = rest % (N / 16);
    int ks = rest / (N / 16);
    int n  = nt * 16 + (lane & 15);
    int kb = ks * 32 + (lane >> 4) * 8;
    unsigned short* o = outp + (size_t)t * 16;
    for (int kk = 0; kk < 16; ++kk)
        o[kk] = f2bfu(W[(size_t)(kb + (kk < 8 ? kk : kk + 8)) * N + n]);
}

// ---------------------------------------------------------------------------
// Main fused kernel: 4 waves/block, 32 edges per wave (M=32 tile: each B
// fragment feeds two WMMAs -> half the B traffic per MAC).
// ---------------------------------------------------------------------------
__global__ __launch_bounds__(128) void relattn_fused(
    const float* __restrict__ node_emb,
    const int*   __restrict__ eidx,     // [2][E]
    const int*   __restrict__ relt,     // [E]
    const float* __restrict__ b_edge,
    const float* __restrict__ b_k,
    const float* __restrict__ b_v,
    const float* __restrict__ b_o1,
    const float* __restrict__ W_o2,
    const float* __restrict__ b_o2,
    const float* __restrict__ qtab,     // [10][128]
    const unsigned short* __restrict__ pWe,   // 512x128 packed
    const unsigned short* __restrict__ pWk,   // 128x128 packed
    const unsigned short* __restrict__ pWv,
    const unsigned short* __restrict__ pWo1,
    float* __restrict__ out, int E)
{
    __shared__ __align__(16) unsigned short abuf[4][32][136]; // padded rows
    __shared__ float sbuf[4][32][4];
    __shared__ int   rbuf[4][32];

    const int tid  = threadIdx.x;
    const int wave = tid >> 5;          // 0..3
    const int l    = tid & 31;
    const int c    = l & 15;
    const int hlf  = l >> 4;

    long base = (long)blockIdx.x * 128 + (long)wave * 32;
    long e0 = base + c;        if (e0 >= E) e0 = E - 1;
    long e1 = base + 16 + c;   if (e1 >= E) e1 = E - 1;
    int s0 = eidx[e0], t0 = eidx[(long)E + e0];
    int s1 = eidx[e1], t1 = eidx[(long)E + e1];
    if (hlf == 0) {
        rbuf[wave][c]      = relt[e0];
        rbuf[wave][16 + c] = relt[e1];
    }

    unsigned short (*ab)[136] = abuf[wave];

    const float* sr0 = node_emb + (size_t)s0 * 256;
    const float* tr0 = node_emb + (size_t)t0 * 256;
    const float* sr1 = node_emb + (size_t)s1 * 256;
    const float* tr1 = node_emb + (size_t)t1 * 256;

    const v8f vzero = {0.f, 0.f, 0.f, 0.f, 0.f, 0.f, 0.f, 0.f};
    v8f acc[2][8];
    for (int m = 0; m < 2; ++m)
        for (int i = 0; i < 8; ++i) acc[m][i] = vzero;

    // ---------------- Stage 1: edge_hidden = relu([src|tgt] @ W_edge + b) ----
    for (int kt = 0; kt < 16; ++kt) {
        int f0 = kt * 32 + (hlf ? 8 : 0);
        int off = (f0 < 256) ? f0 : (f0 - 256);
        const float* rp0 = (f0 < 256) ? (sr0 + off) : (tr0 + off);
        const float* rp1 = (f0 < 256) ? (sr1 + off) : (tr1 + off);
        ABu a0 = mk_afrag_g(rp0);
        ABu a1 = mk_afrag_g(rp1);
        const uint4* bp = (const uint4*)pWe;
        for (int nt = 0; nt < 8; ++nt) {
            ABu b;
            int bi = (((kt << 3) + nt) * 32 + l) * 2;
            b.q[0] = bp[bi];
            b.q[1] = bp[bi + 1];
            acc[0][nt] = WMMA_BF16(a0.v, b.v, acc[0][nt]);
            acc[1][nt] = WMMA_BF16(a1.v, b.v, acc[1][nt]);
        }
    }
    for (int nt = 0; nt < 8; ++nt) {
        int n = nt * 16 + c;
        float bn = b_edge[n];
        for (int m = 0; m < 2; ++m)
            for (int r8 = 0; r8 < 8; ++r8) {
                float v = acc[m][nt][r8] + bn;
                v = v > 0.f ? v : 0.f;
                ab[m * 16 + r8 + hlf * 8][n] = f2bfu(v);
            }
    }
    sbuf[wave][l][0] = 0.f; sbuf[wave][l][1] = 0.f;
    sbuf[wave][l][2] = 0.f; sbuf[wave][l][3] = 0.f;
    __syncthreads();

    // ---------------- Stage 2: K = edge_hidden @ W_k + b_k ; head scores -----
    for (int m = 0; m < 2; ++m)
        for (int i = 0; i < 8; ++i) acc[m][i] = vzero;
    for (int kt = 0; kt < 4; ++kt) {
        int k0 = kt * 32 + (hlf ? 8 : 0);
        ABu a0 = mk_afrag_l(ab[c], k0);
        ABu a1 = mk_afrag_l(ab[16 + c], k0);
        const uint4* bp = (const uint4*)pWk;
        for (int nt = 0; nt < 8; ++nt) {
            ABu b;
            int bi = (((kt << 3) + nt) * 32 + l) * 2;
            b.q[0] = bp[bi];
            b.q[1] = bp[bi + 1];
            acc[0][nt] = WMMA_BF16(a0.v, b.v, acc[0][nt]);
            acc[1][nt] = WMMA_BF16(a1.v, b.v, acc[1][nt]);
        }
    }
    // partial Q.K accumulation across lanes via LDS float atomics (ds_add_f32)
    for (int nt = 0; nt < 8; ++nt) {
        int n = nt * 16 + c;
        float bk = b_k[n];
        int h = nt >> 1;                   // columns 32h..32h+31 -> head h
        for (int m = 0; m < 2; ++m)
            for (int r8 = 0; r8 < 8; ++r8) {
                int M = m * 16 + r8 + hlf * 8;
                float q = qtab[rbuf[wave][M] * 128 + n];
                atomicAdd(&sbuf[wave][M][h], (acc[m][nt][r8] + bk) * q);
            }
    }
    __syncthreads();
    {
        const float scale = 0.17677669529663687f;   // 1/sqrt(HEAD_DIM=32)
        float s0v = sbuf[wave][l][0] * scale;
        float s1v = sbuf[wave][l][1] * scale;
        float s2v = sbuf[wave][l][2] * scale;
        float s3v = sbuf[wave][l][3] * scale;
        float m  = fmaxf(fmaxf(s0v, s1v), fmaxf(s2v, s3v));
        float x0 = __expf(s0v - m), x1 = __expf(s1v - m);
        float x2 = __expf(s2v - m), x3 = __expf(s3v - m);
        float inv = 1.f / (x0 + x1 + x2 + x3);
        sbuf[wave][l][0] = x0 * inv; sbuf[wave][l][1] = x1 * inv;
        sbuf[wave][l][2] = x2 * inv; sbuf[wave][l][3] = x3 * inv;
    }
    __syncthreads();

    // ---------------- Stage 3: V GEMM ; output = attn*V + edge_hidden --------
    for (int m = 0; m < 2; ++m)
        for (int i = 0; i < 8; ++i) acc[m][i] = vzero;
    for (int kt = 0; kt < 4; ++kt) {
        int k0 = kt * 32 + (hlf ? 8 : 0);
        ABu a0 = mk_afrag_l(ab[c], k0);
        ABu a1 = mk_afrag_l(ab[16 + c], k0);
        const uint4* bp = (const uint4*)pWv;
        for (int nt = 0; nt < 8; ++nt) {
            ABu b;
            int bi = (((kt << 3) + nt) * 32 + l) * 2;
            b.q[0] = bp[bi];
            b.q[1] = bp[bi + 1];
            acc[0][nt] = WMMA_BF16(a0.v, b.v, acc[0][nt]);
            acc[1][nt] = WMMA_BF16(a1.v, b.v, acc[1][nt]);
        }
    }
    for (int nt = 0; nt < 8; ++nt) {
        int n = nt * 16 + c;
        float bv = b_v[n];
        int h = nt >> 1;
        for (int m = 0; m < 2; ++m)
            for (int r8 = 0; r8 < 8; ++r8) {
                int M = m * 16 + r8 + hlf * 8;
                float att = sbuf[wave][M][h];
                float eh  = bfu2f(ab[M][n]);
                float o   = att * (acc[m][nt][r8] + bv) + eh;
                ab[M][n]  = f2bfu(o);   // unique owner per (M,n): safe in-place
            }
    }
    __syncthreads();

    // ---------------- Stage 4: h = relu(output @ W_o1 + b_o1) ----------------
    for (int m = 0; m < 2; ++m)
        for (int i = 0; i < 8; ++i) acc[m][i] = vzero;
    for (int kt = 0; kt < 4; ++kt) {
        int k0 = kt * 32 + (hlf ? 8 : 0);
        ABu a0 = mk_afrag_l(ab[c], k0);
        ABu a1 = mk_afrag_l(ab[16 + c], k0);
        const uint4* bp = (const uint4*)pWo1;
        for (int nt = 0; nt < 8; ++nt) {
            ABu b;
            int bi = (((kt << 3) + nt) * 32 + l) * 2;
            b.q[0] = bp[bi];
            b.q[1] = bp[bi + 1];
            acc[0][nt] = WMMA_BF16(a0.v, b.v, acc[0][nt]);
            acc[1][nt] = WMMA_BF16(a1.v, b.v, acc[1][nt]);
        }
    }
    for (int nt = 0; nt < 8; ++nt) {
        int n = nt * 16 + c;
        float bo = b_o1[n];
        for (int m = 0; m < 2; ++m)
            for (int r8 = 0; r8 < 8; ++r8) {
                float v = acc[m][nt][r8] + bo;
                v = v > 0.f ? v : 0.f;
                ab[m * 16 + r8 + hlf * 8][n] = f2bfu(v);
            }
    }
    __syncthreads();

    // ---------------- Stage 5: score = h @ W_o2 + b_o2 (all 32 lanes) --------
    {
        float s = b_o2[0];
        const unsigned short* hrow = ab[l];
        for (int g = 0; g < 16; ++g) {
            uint4 hv = *(const uint4*)(hrow + g * 8);
            const float* wp = W_o2 + g * 8;
            float4 w0 = *(const float4*)wp;
            float4 w1 = *(const float4*)(wp + 4);
            s += bfu2f((unsigned short)(hv.x & 0xffff)) * w0.x;
            s += bfu2f((unsigned short)(hv.x >> 16))    * w0.y;
            s += bfu2f((unsigned short)(hv.y & 0xffff)) * w0.z;
            s += bfu2f((unsigned short)(hv.y >> 16))    * w0.w;
            s += bfu2f((unsigned short)(hv.z & 0xffff)) * w1.x;
            s += bfu2f((unsigned short)(hv.z >> 16))    * w1.y;
            s += bfu2f((unsigned short)(hv.w & 0xffff)) * w1.z;
            s += bfu2f((unsigned short)(hv.w >> 16))    * w1.w;
        }
        long eo = base + l;
        if (eo < E) out[eo] = s;
    }
}

// ---------------------------------------------------------------------------
// Host launcher
// ---------------------------------------------------------------------------
extern "C" void kernel_launch(void* const* d_in, const int* in_sizes, int n_in,
                              void* d_out, int out_size, void* d_ws, size_t ws_size,
                              hipStream_t stream) {
    const float* node    = (const float*)d_in[0];
    const int*   eidx    = (const int*)  d_in[1];
    const int*   relt    = (const int*)  d_in[2];
    const float* rel_emb = (const float*)d_in[3];
    const float* W_edge  = (const float*)d_in[4];
    const float* b_edge  = (const float*)d_in[5];
    const float* W_q     = (const float*)d_in[6];
    const float* b_q     = (const float*)d_in[7];
    const float* W_k     = (const float*)d_in[8];
    const float* b_k     = (const float*)d_in[9];
    const float* W_v     = (const float*)d_in[10];
    const float* b_v     = (const float*)d_in[11];
    const float* W_o1    = (const float*)d_in[12];
    const float* b_o1    = (const float*)d_in[13];
    const float* W_o2    = (const float*)d_in[14];
    const float* b_o2    = (const float*)d_in[15];
    float* out = (float*)d_out;
    int E = in_sizes[2];

    char* ws = (char*)d_ws;
    float*          qtab = (float*)ws;                                  // 5120 B
    unsigned short* pWe  = (unsigned short*)(ws + 8192);                // 128 KB
    unsigned short* pWk  = (unsigned short*)(ws + 8192 + 131072);       // 32 KB
    unsigned short* pWv  = pWk + 128 * 128;
    unsigned short* pWo1 = pWv + 128 * 128;

    build_qtab<<<1, 128, 0, stream>>>(rel_emb, W_q, b_q, qtab);
    pack_b<<<16, 256, 0, stream>>>(W_edge, 16, pWe);   // K=512 -> 16 kTiles
    pack_b<<<4, 256, 0, stream>>>(W_k, 4, pWk);
    pack_b<<<4, 256, 0, stream>>>(W_v, 4, pWv);
    pack_b<<<4, 256, 0, stream>>>(W_o1, 4, pWo1);

    int nb = (E + 127) / 128;
    relattn_fused<<<nb, 128, 0, stream>>>(node, eidx, relt, b_edge, b_k, b_v,
                                          b_o1, W_o2, b_o2, qtab,
                                          pWe, pWk, pWv, pWo1, out, E);
}
